// MultiHeadAttention_52252572123554
// MI455X (gfx1250) — matrix operand order repaired and checked by reading, not compiled, not run on hardware
//
#include <hip/hip_runtime.h>

typedef __bf16 bf16_t;
typedef __attribute__((ext_vector_type(16))) __bf16 v16bf;
typedef __attribute__((ext_vector_type(8)))  float  v8f;
typedef __attribute__((ext_vector_type(4)))  unsigned int u32x4;
typedef __attribute__((ext_vector_type(8)))  int i32x8;
typedef __attribute__((ext_vector_type(4)))  int i32x4;

#define DEVI __device__ __forceinline__

DEVI v8f wmma_bf16(v16bf a, v16bf b, v8f c) {
  // D = A(16x32 bf16) x B(32x16 bf16) + C(16x16 f32)
  return __builtin_amdgcn_wmma_f32_16x16x32_bf16(false, a, false, b, (short)0, c,
                                                 false, false);
}

// A fragment: 16x32 bf16 tile, src row-major, rows contiguous along k.
// ISA layout: lanes 0-15 = rows 0-15 with k{0..7} in v0-3, k{16..23} in v4-7;
//             lanes 16-31 = rows 0-15 with k{8..15} in v0-3, k{24..31} in v4-7.
DEVI v16bf load_a_frag(const bf16_t* base, int ld, int lane) {
  const int r = lane & 15, hv = lane >> 4;
  const bf16_t* p = base + (size_t)r * ld + hv * 8;
  v16bf f;
  reinterpret_cast<uint4*>(&f)[0] = reinterpret_cast<const uint4*>(p)[0];
  reinterpret_cast<uint4*>(&f)[1] = reinterpret_cast<const uint4*>(p + 16)[0];
  return f;
}

// B fragment: 32x16 bf16 tile with B[k][n] = src[n*ld + k] (the 16 output
// columns are rows of src, contiguous along k).
// ISA layout: lanes 0-15 = col n, k{0..7},{16..23}; lanes 16-31 = col n, k{8..15},{24..31}.
DEVI v16bf load_b_frag(const bf16_t* base, int ld, int lane) {
  const int c = lane & 15, hv = lane >> 4;
  const bf16_t* p = base + (size_t)c * ld + hv * 8;
  v16bf f;
  reinterpret_cast<uint4*>(&f)[0] = reinterpret_cast<const uint4*>(p)[0];
  reinterpret_cast<uint4*>(&f)[1] = reinterpret_cast<const uint4*>(p + 16)[0];
  return f;
}

// --------------------------------------------------------------------------
// Tensor Data Mover: async DMA of a 2D tile (tile_k x rows, bf16) from global
// memory into LDS. D# per CDNA5 ISA:
//   group0: count=1 | lds_addr | global_addr(57b) | type=2
//   group1: data_size=2B, pad_enable (pad 4 DWORDs every 16 DWORDs -> LDS row
//           stride = 32 bf16 + 8 bf16 = 40), tensor dims == tile dims (no OOB),
//           tensor_dim0_stride = ld (elements).
// Tracked on TENSORcnt; issue from one wave, s_wait_tensorcnt + barrier.
// --------------------------------------------------------------------------
#define TDM_KCHUNK 32
#define LDS_STRIDE 40  // 32 data + 8 pad bf16 per row

DEVI void tdm_load_tile(unsigned lds_off, const bf16_t* gaddr, int rows, int ld) {
  unsigned long long ga = (unsigned long long)(size_t)gaddr;
  u32x4 g0;
  g0[0] = 1u;                                       // count=1, no gather
  g0[1] = lds_off;                                  // LDS byte offset
  g0[2] = (unsigned)ga;                             // global_addr[31:0]
  g0[3] = ((unsigned)(ga >> 32) & 0x01FFFFFFu) | (2u << 30);  // addr[56:32]|type=2
  const unsigned td0 = (unsigned)TDM_KCHUNK;        // tensor_dim0 == tile_dim0
  const unsigned td1 = (unsigned)rows;              // tensor_dim1 == tile_dim1
  i32x8 g1;
  g1[0] = (int)((1u << 16)      // data_size = 2 bytes
              | (1u << 20)      // pad_enable
              | (3u << 22)      // pad_interval: every 16 DWORDs (64B row)
              | (3u << 25));    // pad_amount: 4 DWORDs (16B)
  g1[1] = (int)((td0 & 0xFFFFu) << 16);                    // tensor_dim0 lo16
  g1[2] = (int)((td0 >> 16) | ((td1 & 0xFFFFu) << 16));    // dim0 hi / dim1 lo
  g1[3] = (int)((td1 >> 16) | ((unsigned)TDM_KCHUNK << 16)); // dim1 hi | tile_dim0
  g1[4] = (int)(unsigned)rows;                             // tile_dim1, tile_dim2=0
  g1[5] = (int)(unsigned)ld;                               // tensor_dim0_stride lo32
  g1[6] = 0;                                               // stride0 hi16, stride1 lo16
  g1[7] = 0;
  i32x4 gz4 = {0, 0, 0, 0};
  i32x8 gz8 = {0, 0, 0, 0, 0, 0, 0, 0};
  __builtin_amdgcn_tensor_load_to_lds(g0, g1, gz4, gz4, gz8, 0);
}

__global__ void cvt_f32_to_bf16(const float* __restrict__ src,
                                bf16_t* __restrict__ dst, int n) {
  int i = blockIdx.x * blockDim.x + threadIdx.x;
  if (i < n) dst[i] = (bf16_t)src[i];
}

// --------------------------------------------------------------------------
// C[i,j] = sum_k A[i,k] * W[j,k] + bias[j]
// A: NxK bf16 row-major, W: MxK bf16 row-major, C: NxM f32.
// Block: 256 threads = 8 waves, output tile 128x64 (wave w -> rows 16w..16w+15,
// all 64 cols). K streamed in 32-wide chunks, double-buffered in LDS by TDM.
// ~42 FLOP per global byte => WMMA-bound, not L2-bound.
// --------------------------------------------------------------------------
#define GTM 128
#define GTN 64
__global__ __launch_bounds__(256) void gemm_nt_bf16_tdm(
    const bf16_t* __restrict__ A, const bf16_t* __restrict__ W,
    const float* __restrict__ bias, float* __restrict__ C,
    int N, int M, int K) {
  __shared__ bf16_t Abuf[2][GTM * LDS_STRIDE];
  __shared__ bf16_t Bbuf[2][GTN * LDS_STRIDE];
  const int lane = threadIdx.x & 31;
  const int wave = threadIdx.x >> 5;
  const int hv = lane >> 4, lc = lane & 15;
  const int tilesN = M / GTN;
  const int ti = blockIdx.x / tilesN;
  const int tj = blockIdx.x % tilesN;
  const bf16_t* Ab = A + (size_t)ti * GTM * K;
  const bf16_t* Wb = W + (size_t)tj * GTN * K;

  if (wave == 0) {
    tdm_load_tile((unsigned)(size_t)&Abuf[0][0], Ab, GTM, K);
    tdm_load_tile((unsigned)(size_t)&Bbuf[0][0], Wb, GTN, K);
    __builtin_amdgcn_s_wait_tensorcnt(0);
  }
  __syncthreads();

  v8f acc0 = {}, acc1 = {}, acc2 = {}, acc3 = {};
  int cur = 0;
  for (int k0 = 0; k0 < K; k0 += TDM_KCHUNK) {
    if (wave == 0 && k0 + TDM_KCHUNK < K) {  // prefetch next chunk into alt buf
      tdm_load_tile((unsigned)(size_t)&Abuf[cur ^ 1][0], Ab + k0 + TDM_KCHUNK, GTM, K);
      tdm_load_tile((unsigned)(size_t)&Bbuf[cur ^ 1][0], Wb + k0 + TDM_KCHUNK, GTN, K);
    }
    const bf16_t* As = &Abuf[cur][0] + (wave * 16) * LDS_STRIDE;
    const bf16_t* Bs = &Bbuf[cur][0];
    const v16bf a = load_a_frag(As, LDS_STRIDE, lane);
    acc0 = wmma_bf16(a, load_b_frag(Bs, LDS_STRIDE, lane), acc0);
    acc1 = wmma_bf16(a, load_b_frag(Bs + 16 * LDS_STRIDE, LDS_STRIDE, lane), acc1);
    acc2 = wmma_bf16(a, load_b_frag(Bs + 32 * LDS_STRIDE, LDS_STRIDE, lane), acc2);
    acc3 = wmma_bf16(a, load_b_frag(Bs + 48 * LDS_STRIDE, LDS_STRIDE, lane), acc3);
    if (wave == 0) __builtin_amdgcn_s_wait_tensorcnt(0);
    __syncthreads();
    cur ^= 1;
  }

  const int col0 = tj * GTN + lc;
  const int rbase = ti * GTM + wave * 16 + hv * 8;
  const float b0 = bias[col0], b1 = bias[col0 + 16];
  const float b2 = bias[col0 + 32], b3 = bias[col0 + 48];
#pragma unroll
  for (int j = 0; j < 8; ++j) {
    float* cr = C + (size_t)(rbase + j) * M;
    cr[col0]      = acc0[j] + b0;
    cr[col0 + 16] = acc1[j] + b1;
    cr[col0 + 32] = acc2[j] + b2;
    cr[col0 + 48] = acc3[j] + b3;
  }
}

// RoPE on Q,K (f32) + repack to per-head bf16. 1/sqrt(dk) folded into Q.
// Qt,Kt: [b,h,s,64] bf16.  Vt: [b,h,64,S] bf16 (transposed for PV B-frags).
__global__ void rope_pack(const float* __restrict__ Qf, const float* __restrict__ Kf,
                          const float* __restrict__ Vf,
                          bf16_t* __restrict__ Qt, bf16_t* __restrict__ Kt,
                          bf16_t* __restrict__ Vt, int B, int S, int H) {
  int idx = blockIdx.x * blockDim.x + threadIdx.x;
  const int i = idx & 31;
  int tmp = idx >> 5;
  const int h = tmp & 15;  // H == 16
  tmp >>= 4;
  const int s = tmp % S;
  const int b = tmp / S;
  if (b >= B) return;
  const int Dm = H * 64;
  const size_t nrow = ((size_t)b * S + s) * Dm + h * 64;
  const float ang = (float)s * __powf(10000.0f, -(float)i * (1.0f / 32.0f));
  const float sn = __sinf(ang), cs = __cosf(ang);
  const float q0 = Qf[nrow + 2 * i], q1 = Qf[nrow + 2 * i + 1];
  const float k0 = Kf[nrow + 2 * i], k1 = Kf[nrow + 2 * i + 1];
  const float qr0 = q0 * cs - q1 * sn, qr1 = q1 * cs + q0 * sn;
  const float kr0 = k0 * cs - k1 * sn, kr1 = k1 * cs + k0 * sn;
  const size_t hoff = ((size_t)(b * H + h) * S + s) * 64;
  const float qscale = 0.125f;  // 1/sqrt(64)
  Qt[hoff + 2 * i]     = (bf16_t)(qr0 * qscale);
  Qt[hoff + 2 * i + 1] = (bf16_t)(qr1 * qscale);
  Kt[hoff + 2 * i]     = (bf16_t)kr0;
  Kt[hoff + 2 * i + 1] = (bf16_t)kr1;
  const float v0 = Vf[nrow + 2 * i], v1 = Vf[nrow + 2 * i + 1];
  const size_t vbase = (size_t)(b * H + h) * 64 * S;
  Vt[vbase + (size_t)(2 * i) * S + s]     = (bf16_t)v0;
  Vt[vbase + (size_t)(2 * i + 1) * S + s] = (bf16_t)v1;
}

// Fused flash-style attention. One wave (32 threads) per (b, h, 16 q-rows).
// 32-key blocks: QK^T (4 WMMA) -> online softmax via LDS -> PV (4 WMMA).
__global__ __launch_bounds__(32) void attn_fused(
    const bf16_t* __restrict__ Qt, const bf16_t* __restrict__ Kt,
    const bf16_t* __restrict__ Vt, const int* __restrict__ mask,
    bf16_t* __restrict__ Ob, int B, int S, int H) {
  const int lane = threadIdx.x & 31;
  const int hv = lane >> 4, lc = lane & 15;
  const int qtiles = S >> 4;
  const int qb = blockIdx.x % qtiles;
  const int bh = blockIdx.x / qtiles;
  const int b = bh / H, h = bh % H;
  const bf16_t* Qh = Qt + (size_t)bh * S * 64;
  const bf16_t* Kh = Kt + (size_t)bh * S * 64;
  const bf16_t* Vh = Vt + (size_t)bh * 64 * S;
  const int* mk = mask + b * S;

  __shared__ float  sc[16 * 36];
  __shared__ bf16_t pb[16 * 40];
  __shared__ float  fbuf[16];

  const int q0 = qb * 16;
  const v16bf qf0 = load_a_frag(Qh + (size_t)q0 * 64, 64, lane);
  const v16bf qf1 = load_a_frag(Qh + (size_t)q0 * 64 + 32, 64, lane);
  v8f o0 = {}, o1 = {}, o2 = {}, o3 = {};
  float m_run = -3.0e38f, l_run = 0.0f;

  for (int j0 = 0; j0 < S; j0 += 32) {
    v8f s0 = {}, s1 = {};
    s0 = wmma_bf16(qf0, load_b_frag(Kh + (size_t)j0 * 64, 64, lane), s0);
    s0 = wmma_bf16(qf1, load_b_frag(Kh + (size_t)j0 * 64 + 32, 64, lane), s0);
    s1 = wmma_bf16(qf0, load_b_frag(Kh + (size_t)(j0 + 16) * 64, 64, lane), s1);
    s1 = wmma_bf16(qf1, load_b_frag(Kh + (size_t)(j0 + 16) * 64 + 32, 64, lane), s1);
    if (mk[j0 + lc] == 0) {
#pragma unroll
      for (int j = 0; j < 8; ++j) s0[j] = -1.0e9f;
    }
    if (mk[j0 + 16 + lc] == 0) {
#pragma unroll
      for (int j = 0; j < 8; ++j) s1[j] = -1.0e9f;
    }
    __syncthreads();
#pragma unroll
    for (int j = 0; j < 8; ++j) {
      sc[(j + hv * 8) * 36 + lc]      = s0[j];
      sc[(j + hv * 8) * 36 + 16 + lc] = s1[j];
    }
    __syncthreads();
    if (lane < 16) {
      float row[32];
      float mx = -3.0e38f;
#pragma unroll
      for (int c = 0; c < 32; ++c) {
        row[c] = sc[lane * 36 + c];
        mx = fmaxf(mx, row[c]);
      }
      const float m_new = fmaxf(m_run, mx);
      const float corr = __expf(m_run - m_new);
      float sum = 0.0f;
#pragma unroll
      for (int c = 0; c < 32; ++c) {
        const float p = __expf(row[c] - m_new);
        sum += p;
        pb[lane * 40 + c] = (bf16_t)p;
      }
      l_run = l_run * corr + sum;
      m_run = m_new;
      fbuf[lane] = corr;
    }
    __syncthreads();
#pragma unroll
    for (int j = 0; j < 8; ++j) {
      const float f = fbuf[j + hv * 8];
      o0[j] *= f; o1[j] *= f; o2[j] *= f; o3[j] *= f;
    }
    v16bf pf;
    {
      const bf16_t* pp = pb + lc * 40 + hv * 8;
      reinterpret_cast<uint4*>(&pf)[0] = reinterpret_cast<const uint4*>(pp)[0];
      reinterpret_cast<uint4*>(&pf)[1] = reinterpret_cast<const uint4*>(pp + 16)[0];
    }
    o0 = wmma_bf16(pf, load_b_frag(Vh + j0, S, lane), o0);
    o1 = wmma_bf16(pf, load_b_frag(Vh + (size_t)16 * S + j0, S, lane), o1);
    o2 = wmma_bf16(pf, load_b_frag(Vh + (size_t)32 * S + j0, S, lane), o2);
    o3 = wmma_bf16(pf, load_b_frag(Vh + (size_t)48 * S + j0, S, lane), o3);
  }
  __syncthreads();
  if (lane < 16) fbuf[lane] = 1.0f / l_run;
  __syncthreads();
  const int Dm = H * 64;
#pragma unroll
  for (int j = 0; j < 8; ++j) {
    const int row = j + hv * 8;
    const float inv = fbuf[row];
    bf16_t* pr = Ob + ((size_t)b * S + q0 + row) * Dm + h * 64 + lc;
    pr[0]  = (bf16_t)(o0[j] * inv);
    pr[16] = (bf16_t)(o1[j] * inv);
    pr[32] = (bf16_t)(o2[j] * inv);
    pr[48] = (bf16_t)(o3[j] * inv);
  }
}

extern "C" void kernel_launch(void* const* d_in, const int* in_sizes, int n_in,
                              void* d_out, int out_size, void* d_ws, size_t ws_size,
                              hipStream_t stream) {
  (void)in_sizes; (void)n_in; (void)out_size; (void)ws_size;
  const int B = 2, S = 2048, D = 1024, H = 16;
  const int N = B * S;
  const float* x  = (const float*)d_in[0];
  const int*   am = (const int*)  d_in[1];
  const float* Wq = (const float*)d_in[2]; const float* bq = (const float*)d_in[3];
  const float* Wk = (const float*)d_in[4]; const float* bk = (const float*)d_in[5];
  const float* Wv = (const float*)d_in[6]; const float* bv = (const float*)d_in[7];
  const float* Wo = (const float*)d_in[8]; const float* bo = (const float*)d_in[9];
  float* out = (float*)d_out;

  char* ws = (char*)d_ws;
  size_t off = 0;
  auto take = [&](size_t bytes) -> char* {
    char* p = ws + off;
    off += (bytes + 255) & ~(size_t)255;
    return p;
  };
  bf16_t* xb  = (bf16_t*)take((size_t)N * D * sizeof(bf16_t));
  bf16_t* Wqb = (bf16_t*)take((size_t)D * D * sizeof(bf16_t));
  bf16_t* Wkb = (bf16_t*)take((size_t)D * D * sizeof(bf16_t));
  bf16_t* Wvb = (bf16_t*)take((size_t)D * D * sizeof(bf16_t));
  bf16_t* Wob = (bf16_t*)take((size_t)D * D * sizeof(bf16_t));
  float*  Qf  = (float*) take((size_t)N * D * sizeof(float));
  float*  Kf  = (float*) take((size_t)N * D * sizeof(float));
  float*  Vf  = (float*) take((size_t)N * D * sizeof(float));
  bf16_t* Qt  = (bf16_t*)take((size_t)N * D * sizeof(bf16_t));
  bf16_t* Kt  = (bf16_t*)take((size_t)N * D * sizeof(bf16_t));
  bf16_t* Vt  = (bf16_t*)take((size_t)N * D * sizeof(bf16_t));
  bf16_t* Ob  = (bf16_t*)take((size_t)N * D * sizeof(bf16_t));

  const int nx = N * D, nw = D * D;
  cvt_f32_to_bf16<<<(nx + 255) / 256, 256, 0, stream>>>(x,  xb,  nx);
  cvt_f32_to_bf16<<<(nw + 255) / 256, 256, 0, stream>>>(Wq, Wqb, nw);
  cvt_f32_to_bf16<<<(nw + 255) / 256, 256, 0, stream>>>(Wk, Wkb, nw);
  cvt_f32_to_bf16<<<(nw + 255) / 256, 256, 0, stream>>>(Wv, Wvb, nw);
  cvt_f32_to_bf16<<<(nw + 255) / 256, 256, 0, stream>>>(Wo, Wob, nw);

  const int gblocks = (N / GTM) * (D / GTN);  // 32 * 16 = 512 blocks
  gemm_nt_bf16_tdm<<<gblocks, 256, 0, stream>>>(xb, Wqb, bq, Qf, N, D, D);
  gemm_nt_bf16_tdm<<<gblocks, 256, 0, stream>>>(xb, Wkb, bk, Kf, N, D, D);
  gemm_nt_bf16_tdm<<<gblocks, 256, 0, stream>>>(xb, Wvb, bv, Vf, N, D, D);

  const int nrope = N * H * 32;
  rope_pack<<<(nrope + 255) / 256, 256, 0, stream>>>(Qf, Kf, Vf, Qt, Kt, Vt, B, S, H);

  attn_fused<<<B * H * (S / 16), 32, 0, stream>>>(Qt, Kt, Vt, am, Ob, B, S, H);

  gemm_nt_bf16_tdm<<<gblocks, 256, 0, stream>>>(Ob, Wob, bo, out, N, D, D);
}
